// PCN_37391985279556
// MI455X (gfx1250) — compile-verified, matrix operand order checked by reading, not run on hardware
//
#include <hip/hip_runtime.h>
#include <hip/hip_bf16.h>

// ---------------- problem constants ----------------
#define NB   4096      // graphs
#define MPTS 64        // points per graph
#define KNN  3
#define FX   59
#define EPS  1e-5f

typedef __attribute__((ext_vector_type(16))) _Float16 v16h;
typedef __attribute__((ext_vector_type(8)))  _Float16 v8h;
typedef __attribute__((ext_vector_type(8)))  float    v8f;

// ---------------- WMMA helpers (ISA 7.12.2 layouts) ----------------
// Load a 16x32 f16 A-fragment (or symmetric 32x16 B-fragment from an [N][K]
// transposed weight array) from a row-major f16 array with `stride` halves/row.
// lane<16 : row=row0+lane,    K = k0+{0..7, 16..23}
// lane>=16: row=row0+lane-16, K = k0+{8..15, 24..31}
__device__ __forceinline__ v16h load_frag16(const _Float16* base, int row0,
                                            int stride, int k0, int lane) {
  int lo = lane & 15, hi = lane >> 4;
  const _Float16* p = base + (size_t)(row0 + lo) * stride + k0 + hi * 8;
  v8h x0 = *(const v8h*)(p);
  v8h x1 = *(const v8h*)(p + 16);
  return __builtin_shufflevector(x0, x1, 0,1,2,3,4,5,6,7,8,9,10,11,12,13,14,15);
}

__device__ __forceinline__ v8f wmma16(v16h a, v16h b, v8f c) {
  // D = A(16x32) * B(32x16) + C, f32 accumulate
  return __builtin_amdgcn_wmma_f32_16x16x32_f16(false, a, false, b,
                                                (short)0, c, false, false);
}

// ---------------- k0: weight transpose/convert + zero BN accumulators ----
// W1T [128][64] (62 padded to 64), W2T [128][128], Wc1T [256][128], bn zeroed.
__global__ void __launch_bounds__(256) k_prep(const float* __restrict__ W1,
                                              const float* __restrict__ W2,
                                              const float* __restrict__ Wc1,
                                              _Float16* __restrict__ W1T,
                                              _Float16* __restrict__ W2T,
                                              _Float16* __restrict__ Wc1T,
                                              float* __restrict__ bnz) {
  int i = blockIdx.x * 256 + threadIdx.x;
  if (i < 128 * 64) {
    int n = i >> 6, k = i & 63;
    W1T[i] = (_Float16)(k < 62 ? W1[k * 128 + n] : 0.f);
  }
  int j = i - 8192;
  if (j >= 0 && j < 128 * 128) {
    int n = j >> 7, k = j & 127;
    W2T[j] = (_Float16)W2[k * 128 + n];
  }
  int l = i - 8192 - 16384;
  if (l >= 0 && l < 256 * 128) {
    int n = l >> 7, k = l & 127;
    Wc1T[l] = (_Float16)Wc1[k * 256 + n];
  }
  int z = i - 8192 - 16384 - 32768;
  if (z >= 0 && z < 1026) bnz[z] = 0.f;   // bn1(256)+bn2(256)+bn3(512)+bn4(2)
}

// ---------------- k1: per-graph kNN + edge MLP GEMM + max aggregate -------
// One block per graph, 128 threads (4 wave32). GEMM [192x64]x[64x128].
// Wave w owns M-tiles 3w..3w+2 (48 rows = node-triples 16w..16w+15), so the
// K-max reduction is wave-local. BN1 sum/sumsq accumulated via LDS atomics.
__global__ void __launch_bounds__(128) k_edge(const float* __restrict__ x,
                                              const float* __restrict__ pos,
                                              const float* __restrict__ b1,
                                              const _Float16* __restrict__ W1T,
                                              float* __restrict__ bn1,
                                              _Float16* __restrict__ m1h) {
  __shared__ __align__(16) float    posS[MPTS][4];
  __shared__ int                    nbrS[MPTS][4];
  __shared__ __align__(16) _Float16 msgS[192][64];
  __shared__ __align__(16) _Float16 stageS[4][48][16];
  __shared__ float sumS[128], sqS[128];

  const int g = blockIdx.x, tid = threadIdx.x;
  const int lane = tid & 31, wv = tid >> 5;
  __builtin_prefetch(W1T + tid * 64, 0, 0);   // warm L2/WGP$ for B-matrix

  if (tid < MPTS) {
    posS[tid][0] = pos[(size_t)(g * MPTS + tid) * 3 + 0];
    posS[tid][1] = pos[(size_t)(g * MPTS + tid) * 3 + 1];
    posS[tid][2] = pos[(size_t)(g * MPTS + tid) * 3 + 2];
  }
  sumS[tid] = 0.f; sqS[tid] = 0.f;
  __syncthreads();

  // ---- kNN: 3 smallest squared distances, self excluded ----
  if (tid < MPTS) {
    float px = posS[tid][0], py = posS[tid][1], pz = posS[tid][2];
    float d0 = 3e38f, d1 = 3e38f, d2 = 3e38f;
    int   i0 = 0, i1 = 0, i2 = 0;
    for (int j = 0; j < MPTS; ++j) {
      if (j == tid) continue;
      float dx = posS[j][0] - px, dy = posS[j][1] - py, dz = posS[j][2] - pz;
      float d = dx * dx + dy * dy + dz * dz;
      if (d < d0)      { d2=d1; i2=i1; d1=d0; i1=i0; d0=d; i0=j; }
      else if (d < d1) { d2=d1; i2=i1; d1=d;  i1=j; }
      else if (d < d2) { d2=d;  i2=j; }
    }
    nbrS[tid][0] = i0; nbrS[tid][1] = i1; nbrS[tid][2] = i2;
  }
  __syncthreads();

  // ---- build edge messages [192][64] f16: [xj(59) | pj-pi(3) | pad(2)] ----
  for (int idx = tid; idx < 192 * 64; idx += 128) {
    int r = idx >> 6, c = idx & 63;
    int i = r / 3, kk = r - i * 3;
    int src = nbrS[i][kk];
    float v;
    if (c < FX)       v = x[(size_t)(g * MPTS + src) * FX + c];
    else if (c < 62)  v = posS[src][c - FX] - posS[i][c - FX];
    else              v = 0.f;
    msgS[r][c] = (_Float16)v;
  }
  __syncthreads();

  // ---- GEMM1 + bias + ReLU + BN stats + per-node max ----
  const int lo = lane & 15, hi = lane >> 4;
  for (int nt = 0; nt < 8; ++nt) {
    const int col = nt * 16 + lo;
    const float bias = b1[col];
    float s = 0.f, q = 0.f;
    for (int m3 = 0; m3 < 3; ++m3) {
      v8f acc = {};
      for (int kt = 0; kt < 2; ++kt) {
        v16h a = load_frag16(&msgS[0][0], (wv * 3 + m3) * 16, 64, kt * 32, lane);
        v16h b = load_frag16(W1T, nt * 16, 64, kt * 32, lane);
        acc = wmma16(a, b, acc);
      }
#pragma unroll
      for (int r = 0; r < 8; ++r) {
        float v = acc[r] + bias;
        v = v > 0.f ? v : 0.f;
        s += v; q += v * v;
        stageS[wv][m3 * 16 + hi * 8 + r][lo] = (_Float16)v;
      }
    }
    atomicAdd(&sumS[col], s);
    atomicAdd(&sqS[col], q);
    __builtin_amdgcn_wave_barrier();
    // wave-local max over each node's 3 edges, write maxed feature (pre-BN)
    for (int it = 0; it < 8; ++it) {
      int idx = it * 32 + lane;
      int nl = idx >> 4, cl = idx & 15;
      float a0 = (float)stageS[wv][nl * 3 + 0][cl];
      float a1 = (float)stageS[wv][nl * 3 + 1][cl];
      float a2 = (float)stageS[wv][nl * 3 + 2][cl];
      float m  = fmaxf(a0, fmaxf(a1, a2));
      int node = g * MPTS + wv * 16 + nl;
      m1h[(size_t)node * 128 + nt * 16 + cl] = (_Float16)m;
    }
    __builtin_amdgcn_wave_barrier();
  }
  __syncthreads();
  atomicAdd(&bn1[tid],       sumS[tid]);   // global channel sums for BN1
  atomicAdd(&bn1[128 + tid], sqS[tid]);
}

// ---------------- k2: node MLP. BN1 folded into A-load; a2 never stored:
// per-graph row-sums feed both BN2 stats and the mean-pool directly. -------
__global__ void __launch_bounds__(128) k_node(const _Float16* __restrict__ m1h,
                                              const float* __restrict__ bn1,
                                              const float* __restrict__ g1,
                                              const float* __restrict__ be1,
                                              const _Float16* __restrict__ W2T,
                                              const float* __restrict__ b2,
                                              float* __restrict__ bn2,
                                              float* __restrict__ pooled_raw) {
  __shared__ __align__(16) _Float16 Ah[64][128];
  __shared__ float sumS[128], sqS[128];
  const int g = blockIdx.x, tid = threadIdx.x;
  const int lane = tid & 31, wv = tid >> 5;
  const float invE = 1.f / (float)(NB * MPTS * KNN);
  __builtin_prefetch(W2T + tid * 128, 0, 0);

  sumS[tid] = 0.f; sqS[tid] = 0.f;
  for (int idx = tid; idx < 64 * 128; idx += 128) {
    int r = idx >> 7, c = idx & 127;
    float mu  = bn1[c] * invE;
    float var = bn1[128 + c] * invE - mu * mu;
    float sc  = g1[c] * rsqrtf(var + EPS);
    float sh  = be1[c] - mu * sc;
    float v = (float)m1h[(size_t)(g * MPTS + r) * 128 + c] * sc + sh;
    Ah[r][c] = (_Float16)v;
  }
  __syncthreads();

  const int lo = lane & 15;
  for (int nt = 0; nt < 8; ++nt) {
    v8f acc = {};
    for (int kt = 0; kt < 4; ++kt) {
      v16h a = load_frag16(&Ah[0][0], wv * 16, 128, kt * 32, lane);
      v16h b = load_frag16(W2T, nt * 16, 128, kt * 32, lane);
      acc = wmma16(a, b, acc);
    }
    const int col = nt * 16 + lo;
    const float bias = b2[col];
    float s = 0.f, q = 0.f;
#pragma unroll
    for (int r = 0; r < 8; ++r) {
      float v = acc[r] + bias;
      v = v > 0.f ? v : 0.f;
      s += v; q += v * v;
    }
    atomicAdd(&sumS[col], s);
    atomicAdd(&sqS[col], q);
  }
  __syncthreads();
  float s = sumS[tid];
  pooled_raw[(size_t)g * 128 + tid] = s;       // per-graph row sum (pre-BN2)
  atomicAdd(&bn2[tid],       s);
  atomicAdd(&bn2[128 + tid], sqS[tid]);
}

// ---------------- k3: classifier hidden layer [4096x128]x[128x256] --------
__global__ void __launch_bounds__(128) k_cls1(const float* __restrict__ pooled_raw,
                                              const float* __restrict__ bn2,
                                              const float* __restrict__ g2,
                                              const float* __restrict__ be2,
                                              const _Float16* __restrict__ Wc1T,
                                              const float* __restrict__ bc1,
                                              float* __restrict__ bn3,
                                              float* __restrict__ c1raw) {
  __shared__ __align__(16) _Float16 Ph[64][128];
  __shared__ float sumS[256], sqS[256];
  const int tid = threadIdx.x, lane = tid & 31, wv = tid >> 5;
  const int row0 = blockIdx.x * 64;
  const float invN = 1.f / (float)(NB * MPTS);
  __builtin_prefetch(Wc1T + tid * 256, 0, 0);

  for (int i = tid; i < 256; i += 128) { sumS[i] = 0.f; sqS[i] = 0.f; }
  for (int idx = tid; idx < 64 * 128; idx += 128) {
    int r = idx >> 7, c = idx & 127;
    float mu  = bn2[c] * invN;
    float var = bn2[128 + c] * invN - mu * mu;
    float sc  = g2[c] * rsqrtf(var + EPS);
    float sh  = be2[c] - mu * sc;
    float v = (pooled_raw[(size_t)(row0 + r) * 128 + c] * (1.f / (float)MPTS)) * sc + sh;
    Ph[r][c] = (_Float16)v;
  }
  __syncthreads();

  const int lo = lane & 15, hi = lane >> 4;
  for (int nt = 0; nt < 16; ++nt) {
    v8f acc = {};
    for (int kt = 0; kt < 4; ++kt) {
      v16h a = load_frag16(&Ph[0][0], wv * 16, 128, kt * 32, lane);
      v16h b = load_frag16(Wc1T, nt * 16, 128, kt * 32, lane);
      acc = wmma16(a, b, acc);
    }
    const int col = nt * 16 + lo;
    const float bias = bc1[col];
    float s = 0.f, q = 0.f;
#pragma unroll
    for (int r = 0; r < 8; ++r) {
      float v = acc[r] + bias;
      v = v > 0.f ? v : 0.f;
      int row = row0 + wv * 16 + hi * 8 + r;
      c1raw[(size_t)row * 256 + col] = v;
      s += v; q += v * v;
    }
    atomicAdd(&sumS[col], s);
    atomicAdd(&sqS[col], q);
  }
  __syncthreads();
  for (int i = tid; i < 256; i += 128) {
    atomicAdd(&bn3[i],       sumS[i]);
    atomicAdd(&bn3[256 + i], sqS[i]);
  }
}

// ---------------- k4: final linear 256->1 (BN3 folded) + BN4 stats --------
__global__ void __launch_bounds__(256) k_cls2(const float* __restrict__ c1raw,
                                              const float* __restrict__ bn3,
                                              const float* __restrict__ gc1,
                                              const float* __restrict__ bec1,
                                              const float* __restrict__ Wc2,
                                              const float* __restrict__ bc2,
                                              float* __restrict__ bn4,
                                              float* __restrict__ c2raw) {
  __shared__ float redS[256], redQ[256];
  const int b = blockIdx.x * 256 + threadIdx.x;
  const float invB = 1.f / (float)NB;
  float acc = 0.f;
  for (int c = 0; c < 256; ++c) {
    float mu  = bn3[c] * invB;
    float var = bn3[256 + c] * invB - mu * mu;
    float sc  = gc1[c] * rsqrtf(var + EPS);
    float sh  = bec1[c] - mu * sc;
    acc += (c1raw[(size_t)b * 256 + c] * sc + sh) * Wc2[c];
  }
  float v = acc + bc2[0];
  v = v > 0.f ? v : 0.f;
  c2raw[b] = v;
  redS[threadIdx.x] = v; redQ[threadIdx.x] = v * v;
  __syncthreads();
  for (int off = 128; off > 0; off >>= 1) {
    if (threadIdx.x < off) {
      redS[threadIdx.x] += redS[threadIdx.x + off];
      redQ[threadIdx.x] += redQ[threadIdx.x + off];
    }
    __syncthreads();
  }
  if (threadIdx.x == 0) {
    atomicAdd(&bn4[0], redS[0]);
    atomicAdd(&bn4[1], redQ[0]);
  }
}

// ---------------- k5: BN4 + sigmoid ---------------------------------------
__global__ void __launch_bounds__(256) k_final(const float* __restrict__ c2raw,
                                               const float* __restrict__ bn4,
                                               const float* __restrict__ gc2,
                                               const float* __restrict__ bec2,
                                               float* __restrict__ out) {
  const int b = blockIdx.x * 256 + threadIdx.x;
  const float invB = 1.f / (float)NB;
  float mu  = bn4[0] * invB;
  float var = bn4[1] * invB - mu * mu;
  float y = gc2[0] * (c2raw[b] - mu) * rsqrtf(var + EPS) + bec2[0];
  out[b] = 1.f / (1.f + expf(-y));
}

// ---------------- workspace layout (bytes) --------------------------------
static const size_t OFF_W1T    = 0;          // 128*64*2   = 16384
static const size_t OFF_W2T    = 16384;      // 128*128*2  = 32768
static const size_t OFF_WC1T   = 49152;      // 256*128*2  = 65536
static const size_t OFF_BN1    = 114688;     // 256 f32
static const size_t OFF_BN2    = 115712;     // 256 f32
static const size_t OFF_BN3    = 116736;     // 512 f32
static const size_t OFF_BN4    = 118784;     // 2 f32 (padded)
static const size_t OFF_POOLED = 119040;     // 4096*128*4 = 2097152
static const size_t OFF_C1RAW  = 2216192;    // 4096*256*4 = 4194304
static const size_t OFF_C2RAW  = 6410496;    // 4096*4
static const size_t OFF_M1H    = 6426880;    // 262144*128*2 = 67108864

extern "C" void kernel_launch(void* const* d_in, const int* in_sizes, int n_in,
                              void* d_out, int out_size, void* d_ws, size_t ws_size,
                              hipStream_t stream) {
  const float* x    = (const float*)d_in[0];
  const float* pos  = (const float*)d_in[1];
  // d_in[2] = batch (layout is known: node/64), unused
  const float* W1   = (const float*)d_in[3];
  const float* b1   = (const float*)d_in[4];
  const float* g1   = (const float*)d_in[5];
  const float* be1  = (const float*)d_in[6];
  const float* W2   = (const float*)d_in[7];
  const float* b2   = (const float*)d_in[8];
  const float* g2   = (const float*)d_in[9];
  const float* be2  = (const float*)d_in[10];
  const float* Wc1  = (const float*)d_in[11];
  const float* bc1  = (const float*)d_in[12];
  const float* gc1  = (const float*)d_in[13];
  const float* bec1 = (const float*)d_in[14];
  const float* Wc2  = (const float*)d_in[15];
  const float* bc2  = (const float*)d_in[16];
  const float* gc2  = (const float*)d_in[17];
  const float* bec2 = (const float*)d_in[18];

  char* ws = (char*)d_ws;
  _Float16* W1T  = (_Float16*)(ws + OFF_W1T);
  _Float16* W2T  = (_Float16*)(ws + OFF_W2T);
  _Float16* Wc1T = (_Float16*)(ws + OFF_WC1T);
  float* bn1     = (float*)(ws + OFF_BN1);
  float* bn2     = (float*)(ws + OFF_BN2);
  float* bn3     = (float*)(ws + OFF_BN3);
  float* bn4     = (float*)(ws + OFF_BN4);
  float* pooled  = (float*)(ws + OFF_POOLED);
  float* c1raw   = (float*)(ws + OFF_C1RAW);
  float* c2raw   = (float*)(ws + OFF_C2RAW);
  _Float16* m1h  = (_Float16*)(ws + OFF_M1H);
  float* out     = (float*)d_out;

  k_prep <<<229, 256, 0, stream>>>(W1, W2, Wc1, W1T, W2T, Wc1T, bn1);
  k_edge <<<NB, 128, 0, stream>>>(x, pos, b1, W1T, bn1, m1h);
  k_node <<<NB, 128, 0, stream>>>(m1h, bn1, g1, be1, W2T, b2, bn2, pooled);
  k_cls1 <<<NB / 64, 128, 0, stream>>>(pooled, bn2, g2, be2, Wc1T, bc1, bn3, c1raw);
  k_cls2 <<<NB / 256, 256, 0, stream>>>(c1raw, bn3, gc1, bec1, Wc2, bc2, bn4, c2raw);
  k_final<<<NB / 256, 256, 0, stream>>>(c2raw, bn4, gc2, bec2, out);
}
